// Synthesizer_62843961475839
// MI455X (gfx1250) — matrix operand order repaired and checked by reading, not compiled, not run on hardware
//
#include <hip/hip_runtime.h>
#include <hip/hip_bf16.h>

// ---------------- problem constants ----------------
constexpr int kB = 16, kT = 400, kE = 768;
constexpr int kP = 256, kH = 1024, kA = 128, kL = 32, kM = 80;
constexpr int kLSTM_IN = 1024;   // P + E
constexpr int kPROJ_IN = 1792;   // H + E
constexpr int NBLK = 16, NTHR = 512;   // persistent decoder grid: 256 waves total

// ---------------- workspace layout (bytes, all 256-aligned) ----------------
// NOTE: LSTMIN16|H0F16 contiguous (phase-1 LDS stage), H0F16|H1F16 contiguous (phase-2 stage).
constexpr size_t OFF_BAR      = 0;
constexpr size_t OFF_LSTMIN16 = 256;                                   // f16 [16][1024]
constexpr size_t OFF_H0F16    = OFF_LSTMIN16 + (size_t)kB*kLSTM_IN*2;  // f16 [16][1024]
constexpr size_t OFF_H1F16    = OFF_H0F16    + (size_t)kB*kH*2;        // f16 [16][1024]
constexpr size_t OFF_C0       = OFF_H1F16    + (size_t)kB*kH*2;        // f32 [16][1024]
constexpr size_t OFF_C1       = OFF_C0       + (size_t)kB*kH*4;
constexpr size_t OFF_H1       = OFF_C1       + (size_t)kB*kH*4;        // f32 (read by proj/Ws)
constexpr size_t OFF_ATT      = OFF_H1       + (size_t)kB*kH*4;        // step_att f32 [16][768]
constexpr size_t OFF_STEPIN   = OFF_ATT      + (size_t)kB*kE*4;        // f32 [16][80]
constexpr size_t OFF_CUM      = OFF_STEPIN   + (size_t)kB*kM*4;        // f32 [16][400]
constexpr size_t OFF_AW       = OFF_CUM      + (size_t)kB*kT*4;
constexpr size_t OFF_ENG      = OFF_AW       + (size_t)kB*kT*4;
constexpr size_t OFF_WSATT    = OFF_ENG      + (size_t)kB*kT*4;        // Ws f32 [16][128]
constexpr size_t STATE_END    = OFF_WSATT    + (size_t)kB*kA*4;        // zeroed each call
constexpr size_t OFF_WIH0H    = STATE_END;                             // f16 [4096][1024]
constexpr size_t OFF_WHH0H    = OFF_WIH0H + (size_t)4096*kH*2;
constexpr size_t OFF_WIH1H    = OFF_WHH0H + (size_t)4096*kH*2;
constexpr size_t OFF_WHH1H    = OFF_WIH1H + (size_t)4096*kH*2;
constexpr size_t OFF_W2K      = OFF_WHH1H + (size_t)4096*kH*2;         // f16 [5][256][256]
constexpr size_t OFF_W3K      = OFF_W2K   + (size_t)5*256*256*2;
constexpr size_t OFF_VH       = OFF_W3K   + (size_t)5*256*256*2;       // f32 [16][400][128]
constexpr size_t OFF_MEL      = OFF_VH    + (size_t)kB*kT*kA*4;        // f32 [16][400][80]
constexpr size_t WS_NEED      = OFF_MEL   + (size_t)kB*kT*kM*4;        // ~40 MB

// ---------------- WMMA helpers ----------------
typedef _Float16 half8  __attribute__((ext_vector_type(8)));
typedef _Float16 half16 __attribute__((ext_vector_type(16)));
typedef float    f32x8  __attribute__((ext_vector_type(8)));
union H16 { half16 v; half8 h[2]; };

__device__ __forceinline__ f32x8 wmma16(half16 a, half16 b, f32x8 c) {
  return __builtin_amdgcn_wmma_f32_16x16x32_f16(false, a, false, b, (short)0, c, false, false);
}
__device__ __forceinline__ float sigm(float x) { return 1.f / (1.f + __expf(-x)); }

// software grid barrier (persistent kernel; 16 blocks trivially co-resident on MI455X)
__device__ __forceinline__ void grid_sync(unsigned* bar, unsigned nblk, unsigned& gen) {
  __syncthreads();
  gen += nblk;
  if (threadIdx.x == 0) {
    __threadfence();
    atomicAdd(bar, 1u);
    while (__atomic_load_n(bar, __ATOMIC_RELAXED) < gen) __builtin_amdgcn_s_sleep(2);
    __threadfence();
  }
  __syncthreads();
}

// Gate GEMM, block-local tiling: block owns 64 hidden cols x 4 gates; its 16 waves
// split K 4-ways (wave = group gib 0..3, quarter q 0..3). A operands come from LDS
// (staged once per phase), weights stream global->VGPR (no reuse -> direct path).
// Partials land in LDS (reusing the stage area) for a __syncthreads-only reduction.
__device__ __forceinline__ void lstm_gate_gemm_q(
    const _Float16* a1, const _Float16* a2,                 // LDS [16][1024] each
    const _Float16* __restrict__ w1, const _Float16* __restrict__ w2, // global [4096][1024]
    float* lredux, int warp, int n0, int q, int lane)
{
  const int row  = lane & 15;
  const int aoff = (lane < 16) ? 0 : 16;   // A half-wave K offset (contiguous, matches B)
  const int koff = (lane < 16) ? 0 : 16;   // B half-wave K offset (contiguous k = 16h + s, matches A)
  const int kq   = q * 256;
  const _Float16* a1p = a1 + (size_t)row * kH + kq + aoff;
  const _Float16* a2p = a2 + (size_t)row * kH + kq + aoff;
  const int brow = (lane < 16) ? 0 : 8;    // ISA D-tile: lanes>=16 -> M=r+8
  #pragma unroll
  for (int gate = 0; gate < 4; ++gate) {
    const int nrow = gate * kH + n0 + row;
    const _Float16* w1p = w1 + (size_t)nrow * kH + kq + koff;
    const _Float16* w2p = w2 + (size_t)nrow * kH + kq + koff;
    __builtin_prefetch(w2p, 0, 1);
    f32x8 acc = {};
    for (int k = 0; k < 256; k += 32) {
      H16 a; a.h[0] = *(const half8*)(a1p + k); a.h[1] = *(const half8*)(a1p + k + 8);
      acc = wmma16(a.v, *(const half16*)(w1p + k), acc);
    }
    for (int k = 0; k < 256; k += 32) {
      H16 a; a.h[0] = *(const half8*)(a2p + k); a.h[1] = *(const half8*)(a2p + k + 8);
      acc = wmma16(a.v, *(const half16*)(w2p + k), acc);
    }
    float* dst = lredux + (size_t)((warp * 4 + gate) << 8) + (brow << 4) + row;
    #pragma unroll
    for (int r = 0; r < 8; ++r) dst[r * 16] = acc[r];
  }
}

// 4-way K reduction + i,f,g,o combine, all block-local (1024 (b,n) elems / block)
__device__ __forceinline__ void gate_combine_block(
    const float* lredux, const float* __restrict__ bih, const float* __restrict__ bhh,
    float* __restrict__ c, float* __restrict__ hf32 /*nullable*/,
    _Float16* __restrict__ hf16, int tid, int blk)
{
  for (int i = tid; i < kB * 64; i += NTHR) {
    int b = i >> 6, nl = i & 63;
    int gib = nl >> 4, ncol = nl & 15;
    float z[4];
    #pragma unroll
    for (int g = 0; g < 4; ++g) {
      float s = 0.f;
      #pragma unroll
      for (int q = 0; q < 4; ++q)
        s += lredux[(size_t)(((gib * 4 + q) * 4 + g) << 8) + (b << 4) + ncol];
      z[g] = s;
    }
    int ng = blk * 64 + nl;
    float zi = z[0] + bih[ng]        + bhh[ng];
    float zf = z[1] + bih[ng + 1024] + bhh[ng + 1024];
    float zg = z[2] + bih[ng + 2048] + bhh[ng + 2048];
    float zo = z[3] + bih[ng + 3072] + bhh[ng + 3072];
    int gi = b * kH + ng;
    float cc = sigm(zf) * c[gi] + sigm(zi) * tanhf(zg);
    float hh = sigm(zo) * tanhf(cc);
    c[gi] = cc;
    if (hf32) hf32[gi] = hh;
    hf16[gi] = (_Float16)hh;
  }
}

// ---------------- prep kernels ----------------
__global__ void f32_to_f16_kernel(const float* __restrict__ src, _Float16* __restrict__ dst, int n) {
  for (int i = blockIdx.x * blockDim.x + threadIdx.x; i < n; i += gridDim.x * blockDim.x)
    dst[i] = (_Float16)src[i];
}
// pn_w (256,256,1,5) OIHW  ->  per-tap A layout [k][co][ci] f16
__global__ void build_wk_kernel(const float* __restrict__ w, _Float16* __restrict__ dst) {
  for (int i = blockIdx.x * blockDim.x + threadIdx.x; i < 5 * 256 * 256; i += gridDim.x * blockDim.x) {
    int k = i / 65536, r = i % 65536, co = r >> 8, ci = r & 255;
    dst[i] = (_Float16)w[(size_t)(co * 256 + ci) * 5 + k];
  }
}
// Vh[b][t][a] = sum_e x[b][t][e] * V[a][e]
__global__ void vh_kernel(const float* __restrict__ x, const float* __restrict__ V, float* __restrict__ vh) {
  for (int i = blockIdx.x * blockDim.x + threadIdx.x; i < kB * kT * kA; i += gridDim.x * blockDim.x) {
    int a = i % kA, bt = i / kA;
    const float* xr = x + (size_t)bt * kE;
    const float* vr = V + (size_t)a * kE;
    float s = 0.f;
    for (int k = 0; k < kE; ++k) s += xr[k] * vr[k];
    vh[i] = s;
  }
}

// ---------------- persistent fused decoder (6 grid barriers / step) ----------------
__global__ void __launch_bounds__(NTHR)
decoder_kernel(char* __restrict__ ws, const float* __restrict__ x,
  const float* __restrict__ pw1, const float* __restrict__ pb1,
  const float* __restrict__ pw2, const float* __restrict__ pb2,
  const float* __restrict__ bih0, const float* __restrict__ bhh0,
  const float* __restrict__ bih1, const float* __restrict__ bhh1,
  const float* __restrict__ attWw, const float* __restrict__ attWb,
  const float* __restrict__ attU, const float* __restrict__ attF,
  const float* __restrict__ attv,
  const float* __restrict__ projw, const float* __restrict__ projb)
{
  extern __shared__ char dsm[];                 // 64KB: A-stage / K-reduction / prenet tmp
  _Float16* aLDS   = (_Float16*)dsm;            // [16][1024] + [16][1024]
  float*    lredux = (float*)dsm;               // 16 waves x 4 gates x 256 f32 = 64KB
  float*    p1L    = (float*)dsm;               // block 1, softmax phase: [16][256]

  unsigned* bar   = (unsigned*)(ws + OFF_BAR);
  _Float16* lstmin= (_Float16*)(ws + OFF_LSTMIN16);
  _Float16* h0f   = (_Float16*)(ws + OFF_H0F16);
  _Float16* h1f   = (_Float16*)(ws + OFF_H1F16);
  float* c0  = (float*)(ws + OFF_C0);   float* c1 = (float*)(ws + OFF_C1);
  float* h1  = (float*)(ws + OFF_H1);
  float* att = (float*)(ws + OFF_ATT);  float* stepin = (float*)(ws + OFF_STEPIN);
  float* cum = (float*)(ws + OFF_CUM);  float* aw  = (float*)(ws + OFF_AW);
  float* eng = (float*)(ws + OFF_ENG);  float* wsb = (float*)(ws + OFF_WSATT);
  const _Float16* wih0h = (const _Float16*)(ws + OFF_WIH0H);
  const _Float16* whh0h = (const _Float16*)(ws + OFF_WHH0H);
  const _Float16* wih1h = (const _Float16*)(ws + OFF_WIH1H);
  const _Float16* whh1h = (const _Float16*)(ws + OFF_WHH1H);
  const float* vh  = (const float*)(ws + OFF_VH);
  float* mel = (float*)(ws + OFF_MEL);

  const int tid  = threadIdx.x;
  const int gt   = blockIdx.x * NTHR + tid;     // 0..8191
  const int lane = tid & 31;
  const int warp = tid >> 5;                    // 0..15
  const int gib  = warp >> 2;                   // hidden-tile group in block (0..3)
  const int q    = warp & 3;                    // K quarter (0..3)
  const int n0   = (blockIdx.x * 4 + gib) * 16; // hidden col base (0..1008)
  unsigned gen = 0;

  for (int t = 0; t < kT; ++t) {
    // ---- P1: LSTM0  z = lstm_in@wih0^T + h0_prev@whh0^T -> h0,c0 (block-local reduce)
    {
      const uint4* src = (const uint4*)(ws + OFF_LSTMIN16);   // lstmin | h0f contiguous
      uint4* dst = (uint4*)dsm;
      for (int j = tid; j < 4096; j += NTHR) dst[j] = src[j]; // 64KB stage
    }
    __syncthreads();
    lstm_gate_gemm_q(aLDS, aLDS + kB * kH, wih0h, whh0h, lredux, warp, n0, q, lane);
    __syncthreads();   // (A reads done; lredux writes landed via gemm's ds stores)
    gate_combine_block(lredux, bih0, bhh0, c0, nullptr, h0f, tid, blockIdx.x);
    grid_sync(bar, NBLK, gen);

    // ---- P2: LSTM1 -> h1,c1 (decoder state)
    {
      const uint4* src = (const uint4*)(ws + OFF_H0F16);      // h0f | h1f contiguous
      uint4* dst = (uint4*)dsm;
      for (int j = tid; j < 4096; j += NTHR) dst[j] = src[j];
    }
    __syncthreads();
    lstm_gate_gemm_q(aLDS, aLDS + kB * kH, wih1h, whh1h, lredux, warp, n0, q, lane);
    __syncthreads();
    gate_combine_block(lredux, bih1, bhh1, c1, h1, h1f, tid, blockIdx.x);
    grid_sync(bar, NBLK, gen);

    // ---- F: mel projection (stop head dropped: unused by output) + Ws = dec@attW^T
    for (int i = gt; i < kB * kM + kB * kA; i += NBLK * NTHR) {
      if (i < kB * kM) {
        int b = i / kM, m = i % kM;
        const float* wr = projw + (size_t)m * kPROJ_IN;
        const float* hb = h1 + (size_t)b * kH;
        const float* ab = att + (size_t)b * kE;
        float s = projb[m];
        for (int j = 0; j < kH; ++j) s += hb[j] * wr[j];
        for (int j = 0; j < kE; ++j) s += ab[j] * wr[kH + j];
        mel[(size_t)(b * kT + t) * kM + m] = s;
        stepin[b * kM + m] = s;
      } else {
        int j = i - kB * kM; int b = j >> 7, a = j & 127;
        const float* wr = attWw + (size_t)a * kH;
        const float* hb = h1 + (size_t)b * kH;
        float s = attWb[a];
        for (int k = 0; k < kH; ++k) s += hb[k] * wr[k];
        wsb[b * kA + a] = s;
      }
    }
    grid_sync(bar, NBLK, gen);

    // ---- F2: location conv + energies  e[b][t'] = v . tanh(Ws + Vh + U@loc)
    for (int i = gt; i < kB * kT; i += NBLK * NTHR) {
      int b = i / kT, tt = i % kT;
      float loc[kL];
      #pragma unroll
      for (int l = 0; l < kL; ++l) loc[l] = 0.f;
      for (int k = 0; k < 31; ++k) {
        int ts = tt + k - 15;
        if (ts >= 0 && ts < kT) {
          float cv = cum[b * kT + ts];
          #pragma unroll
          for (int l = 0; l < kL; ++l) loc[l] += cv * attF[l * 31 + k];
        }
      }
      const float* wsrow = wsb + b * kA;
      const float* vhrow = vh + (size_t)i * kA;
      float e = 0.f;
      for (int a = 0; a < kA; ++a) {
        float s = wsrow[a] + vhrow[a];
        const float* ur = attU + a * kL;
        #pragma unroll
        for (int l = 0; l < kL; ++l) s += loc[l] * ur[l];
        e += attv[a] * tanhf(s);
      }
      eng[i] = e;
    }
    grid_sync(bar, NBLK, gen);

    // ---- G1: block 0: softmax over T + cum update; block 1: PreNet (p1 in LDS, p2->lstm_in)
    if (blockIdx.x == 0) {
      int w = warp;
      if (w < kB) {
        const float* er = eng + w * kT;
        float mx = -1e30f;
        for (int tt = lane; tt < kT; tt += 32) mx = fmaxf(mx, er[tt]);
        #pragma unroll
        for (int off = 16; off > 0; off >>= 1) mx = fmaxf(mx, __shfl_xor(mx, off, 32));
        float s = 0.f;
        for (int tt = lane; tt < kT; tt += 32) s += __expf(er[tt] - mx);
        #pragma unroll
        for (int off = 16; off > 0; off >>= 1) s += __shfl_xor(s, off, 32);
        float inv = 1.f / s;
        for (int tt = lane; tt < kT; tt += 32) {
          float a = __expf(er[tt] - mx) * inv;
          aw[w * kT + tt] = a;
          cum[w * kT + tt] += a;
        }
      }
    } else if (blockIdx.x == 1) {
      for (int i = tid; i < kB * kP; i += NTHR) {          // PreNet layer1 -> LDS
        int b = i >> 8, o = i & 255;
        const float* wr = pw1 + (size_t)o * kM;
        const float* sr = stepin + b * kM;
        float s = pb1[o];
        for (int m = 0; m < kM; ++m) s += sr[m] * wr[m];
        p1L[i] = fmaxf(s, 0.f);
      }
      __syncthreads();
      for (int i = tid; i < kB * kP; i += NTHR) {          // PreNet layer2 -> lstm_in f16
        int b = i >> 8, o = i & 255;
        const float* wr = pw2 + (size_t)o * kP;
        const float* pr = p1L + b * kP;
        float s = pb2[o];
        for (int j = 0; j < kP; ++j) s += pr[j] * wr[j];
        lstmin[(size_t)b * kLSTM_IN + o] = (_Float16)fmaxf(s, 0.f);
      }
    }
    grid_sync(bar, NBLK, gen);

    // ---- G2: ctx = aw@x -> step_att (f32) AND f16 pack into lstm_in[256:1024]
    for (int i = gt; i < kB * kE; i += NBLK * NTHR) {
      int b = i / kE, e = i % kE;
      const float* ar = aw + b * kT;
      const float* xr = x + (size_t)b * kT * kE + e;
      float s = 0.f;
      for (int tt = 0; tt < kT; ++tt) s += ar[tt] * xr[(size_t)tt * kE];
      att[i] = s;
      lstmin[(size_t)b * kLSTM_IN + kP + e] = (_Float16)s;
    }
    grid_sync(bar, NBLK, gen);
  }
}

// ---------------- fused 4-layer PostNet: one block per (b,t) slice ----------------
// act tiles live in LDS transposed [m_padded][ci] so WMMA B-loads are contiguous.
__device__ __forceinline__ void conv_wmma_layer(
    const _Float16* __restrict__ Tin, _Float16* __restrict__ Tout,
    const _Float16* __restrict__ Wk, const float* __restrict__ bias)
{
  const int wvl  = threadIdx.x >> 5;          // 0..15 == co-tile
  const int lane = threadIdx.x & 31;
  const int row  = lane & 15;
  const int aoff = (lane < 16) ? 0 : 16;
  const int koff = (lane < 16) ? 0 : 16;
  const int co0  = wvl * 16;
  for (int mt = 0; mt < 5; ++mt) {            // 80 = 5 * 16 m-positions
    f32x8 acc = {};
    for (int k = 0; k < 5; ++k) {             // 5 taps -> 5 shifted 256x256 GEMMs
      const _Float16* Arow = Wk + ((size_t)k * 256 + co0 + row) * 256;
      const _Float16* Brow = Tin + (size_t)(mt * 16 + row + k) * 256 + koff;
      for (int k0 = 0; k0 < 256; k0 += 32) {
        H16 a; a.h[0] = *(const half8*)(Arow + k0 + aoff);
               a.h[1] = *(const half8*)(Arow + k0 + 8 + aoff);
        half16 bv = *(const half16*)(Brow + k0);
        acc = wmma16(a.v, bv, acc);
      }
    }
    int cobase = co0 + ((lane < 16) ? 0 : 8);
    int m = mt * 16 + row;
    #pragma unroll
    for (int r = 0; r < 8; ++r) {
      float v = tanhf(acc[r] + bias[cobase + r]);
      Tout[(size_t)(m + 2) * 256 + (cobase + r)] = (_Float16)v;
    }
  }
}

__global__ void __launch_bounds__(NTHR)
postnet_kernel(const char* __restrict__ ws,
               const float* __restrict__ pnw1, const float* __restrict__ pnb1,
               const float* __restrict__ pnb2, const float* __restrict__ pnb3,
               const float* __restrict__ pnw4, const float* __restrict__ pnb4,
               float* __restrict__ out)
{
  extern __shared__ _Float16 smem[];          // 2 x [88][256] f16 (halo rows stay zero)
  _Float16* T0 = smem;
  _Float16* T1 = smem + 88 * 256;
  const float* mel = (const float*)(ws + OFF_MEL);
  const _Float16* W2K = (const _Float16*)(ws + OFF_W2K);
  const _Float16* W3K = (const _Float16*)(ws + OFF_W3K);
  const int bt = blockIdx.x;
  const float* melrow = mel + (size_t)bt * kM;
  const int tid = threadIdx.x;

  for (int i = tid; i < 2 * 88 * 256; i += NTHR) smem[i] = (_Float16)0.f;
  __syncthreads();
  // layer1 (C_in = 1): VALU, write transposed + tanh
  for (int i = tid; i < 256 * kM; i += NTHR) {
    int co = i / kM, m = i % kM;
    float s = pnb1[co];
    #pragma unroll
    for (int k = 0; k < 5; ++k) {
      int mm = m + k - 2;
      float xv = (mm >= 0 && mm < kM) ? melrow[mm] : 0.f;
      s += pnw1[co * 5 + k] * xv;
    }
    T0[(size_t)(m + 2) * 256 + co] = (_Float16)tanhf(s);
  }
  __syncthreads();
  conv_wmma_layer(T0, T1, W2K, pnb2);   // layer2 (WMMA)
  __syncthreads();
  conv_wmma_layer(T1, T0, W3K, pnb3);   // layer3 (WMMA)
  __syncthreads();
  // layer4 (C_out = 1) + residual
  for (int m = tid; m < kM; m += NTHR) {
    float s = pnb4[0];
    for (int ci = 0; ci < 256; ++ci) {
      #pragma unroll
      for (int k = 0; k < 5; ++k)
        s += pnw4[ci * 5 + k] * (float)T0[(size_t)(m + k) * 256 + ci];
    }
    out[(size_t)bt * kM + m] = s + melrow[m];
  }
}

// ---------------- host launch ----------------
extern "C" void kernel_launch(void* const* d_in, const int* in_sizes, int n_in,
                              void* d_out, int out_size, void* d_ws, size_t ws_size,
                              hipStream_t stream) {
  (void)in_sizes; (void)n_in; (void)out_size; (void)ws_size;
  char* ws = (char*)d_ws;
  const float* x     = (const float*)d_in[0];
  const float* pw1   = (const float*)d_in[1];
  const float* pb1   = (const float*)d_in[2];
  const float* pw2   = (const float*)d_in[3];
  const float* pb2   = (const float*)d_in[4];
  const float* wih0  = (const float*)d_in[5];
  const float* whh0  = (const float*)d_in[6];
  const float* bih0  = (const float*)d_in[7];
  const float* bhh0  = (const float*)d_in[8];
  const float* wih1  = (const float*)d_in[9];
  const float* whh1  = (const float*)d_in[10];
  const float* bih1  = (const float*)d_in[11];
  const float* bhh1  = (const float*)d_in[12];
  const float* attWw = (const float*)d_in[13];
  const float* attWb = (const float*)d_in[14];
  const float* attV  = (const float*)d_in[15];
  const float* attU  = (const float*)d_in[16];
  const float* attF  = (const float*)d_in[17];
  const float* attv  = (const float*)d_in[18];
  const float* projw = (const float*)d_in[19];
  const float* projb = (const float*)d_in[20];
  const float* pnw1  = (const float*)d_in[23];
  const float* pnb1  = (const float*)d_in[24];
  const float* pnw2  = (const float*)d_in[25];
  const float* pnb2  = (const float*)d_in[26];
  const float* pnw3  = (const float*)d_in[27];
  const float* pnb3  = (const float*)d_in[28];
  const float* pnw4  = (const float*)d_in[29];
  const float* pnb4  = (const float*)d_in[30];

  // allow >64KB dynamic LDS launches (idempotent; not a stream op, capture-safe)
  hipFuncSetAttribute((const void*)decoder_kernel,
                      hipFuncAttributeMaxDynamicSharedMemorySize, 65536);
  hipFuncSetAttribute((const void*)postnet_kernel,
                      hipFuncAttributeMaxDynamicSharedMemorySize, 2 * 88 * 256 * (int)sizeof(_Float16));

  // zero barrier + recurrent state each call (capture-legal stream op)
  hipMemsetAsync(ws, 0, STATE_END, stream);

  // prep: f16 LSTM weights, per-tap PostNet weights, Vh cache
  f32_to_f16_kernel<<<4096, 256, 0, stream>>>(wih0, (_Float16*)(ws + OFF_WIH0H), 4096 * kH);
  f32_to_f16_kernel<<<4096, 256, 0, stream>>>(whh0, (_Float16*)(ws + OFF_WHH0H), 4096 * kH);
  f32_to_f16_kernel<<<4096, 256, 0, stream>>>(wih1, (_Float16*)(ws + OFF_WIH1H), 4096 * kH);
  f32_to_f16_kernel<<<4096, 256, 0, stream>>>(whh1, (_Float16*)(ws + OFF_WHH1H), 4096 * kH);
  build_wk_kernel<<<640, 512, 0, stream>>>(pnw2, (_Float16*)(ws + OFF_W2K));
  build_wk_kernel<<<640, 512, 0, stream>>>(pnw3, (_Float16*)(ws + OFF_W3K));
  vh_kernel<<<1600, 512, 0, stream>>>(x, attV, (float*)(ws + OFF_VH));

  // persistent fused decoder scan (T=400 steps, 6 grid barriers per step)
  decoder_kernel<<<NBLK, NTHR, 65536, stream>>>(ws, x, pw1, pb1, pw2, pb2,
      bih0, bhh0, bih1, bhh1, attWw, attWb, attU, attF, attv, projw, projb);

  // fused PostNet: one block per (b,t), dual 45KB LDS act tiles
  postnet_kernel<<<kB * kT, NTHR, 2 * 88 * 256 * (int)sizeof(_Float16), stream>>>(
      ws, pnw1, pnb1, pnb2, pnb3, pnw4, pnb4, (float*)d_out);
}
